// JointNet_77541339562203
// MI455X (gfx1250) — matrix operand order repaired and checked by reading, not compiled, not run on hardware
//
#include <hip/hip_runtime.h>
#include <hip/hip_bf16.h>
#include <math.h>

typedef __attribute__((ext_vector_type(16))) __bf16 v16bf;
typedef __attribute__((ext_vector_type(8)))  __bf16 v8bf;
typedef __attribute__((ext_vector_type(4)))  __bf16 v4bf;
typedef __attribute__((ext_vector_type(8)))  float  v8f;

// Problem sizes (fixed by the reference)
constexpr int kB = 8, kT = 256, kU = 64, kD = 512, kH = 512, kV = 512;

// LDS tile geometry
constexpr int kHPad = 520;   // 64 h-rows of 512 bf16 + 8 pad
constexpr int kKC   = 64;    // K elements staged per pipeline step
constexpr int kBPad = 72;    // 256 w2-rows of 64 bf16 + 8 pad
constexpr int kNT   = 256;   // V columns per workgroup

// ---------------------------------------------------------------------------
// Prep 1: E1[bt,h] = enc[bt,:]·w1[h,:] + b1[h]   (bt in [0, B*T))
//         P1[bu,h] = pred[bu,:]·w1[h,:]          (bu in [0, B*U))
// ---------------------------------------------------------------------------
__global__ __launch_bounds__(256) void jn_prep_proj(
    const float* __restrict__ enc, const float* __restrict__ pred,
    const float* __restrict__ w1,  const float* __restrict__ b1,
    float* __restrict__ E1, float* __restrict__ P1)
{
    int gid = blockIdx.x * 256 + threadIdx.x;
    int row = gid >> 9;          // 0 .. (B*T + B*U - 1)
    int h   = gid & 511;
    const float* x = (row < kB * kT) ? (enc + (size_t)row * kD)
                                     : (pred + (size_t)(row - kB * kT) * kD);
    const float4* xv = (const float4*)x;
    const float4* wv = (const float4*)(w1 + (size_t)h * kD);
    float s = 0.f;
#pragma unroll 8
    for (int i = 0; i < kD / 4; ++i) {
        float4 a = xv[i], b = wv[i];
        s += a.x * b.x + a.y * b.y + a.z * b.z + a.w * b.w;
    }
    if (row < kB * kT) E1[(size_t)row * kH + h] = s + b1[h];
    else               P1[(size_t)(row - kB * kT) * kH + h] = s;
}

// ---------------------------------------------------------------------------
// Prep 2: w2 (fp32) -> bf16 copy in workspace (512 KB, stays L2-hot).
// ---------------------------------------------------------------------------
__global__ __launch_bounds__(256) void jn_cvt_w2(
    const float* __restrict__ w2, __bf16* __restrict__ w2bf)
{
    int i = blockIdx.x * 256 + threadIdx.x;
    w2bf[i] = (__bf16)w2[i];
}

// ---------------------------------------------------------------------------
// Main: logits[bt, u, v] = tanh(E1[bt]+P1[b,u]) @ w2^T + b2
// Block = 256 thr (8 waves). Block tile: M=64 (all u), N=256, K=512.
// Wave (mi,nj): mi=wave&3 -> 16 u-rows; nj=wave>>2 -> 128 v-cols (8 n-tiles).
// Software-pipelined: next W2 K-chunk is fetched to registers while the
// current chunk's 16 WMMAs execute from LDS.
// ---------------------------------------------------------------------------
__global__ __launch_bounds__(256) void jn_main(
    const float* __restrict__ E1, const float* __restrict__ P1,
    const __bf16* __restrict__ w2bf, const float* __restrict__ b2,
    float* __restrict__ out)
{
    extern __shared__ char smem[];
    __bf16* hA = (__bf16*)smem;                                          // [64][kHPad]
    __bf16* bL = (__bf16*)(smem + (size_t)kU * kHPad * sizeof(__bf16));  // [256][kBPad]

    const int bt   = blockIdx.x;           // 0..B*T-1
    const int b    = bt >> 8;              // bt / T
    const int n0   = blockIdx.y * kNT;
    const int tid  = threadIdx.x;
    const int lane = tid & 31;
    const int wave = tid >> 5;
    const int mi   = wave & 3;
    const int nj   = wave >> 2;
    const int lrow = lane & 15;
    const bool lo  = lane < 16;

    // ---- Phase 1: h tile = tanh(E1[bt] + P1[b,u]) -> LDS (bf16) ----
    const float* e = E1 + (size_t)bt * kH;
    const float* p = P1 + (size_t)b * kU * kH;
    for (int i = tid; i < kU * (kH / 4); i += 256) {
        int u  = i >> 7;
        int k4 = (i & 127) << 2;
        float4 ev = *(const float4*)(e + k4);
        float4 pv = *(const float4*)(p + (size_t)u * kH + k4);
        v4bf hv;
        hv[0] = (__bf16)tanhf(ev.x + pv.x);
        hv[1] = (__bf16)tanhf(ev.y + pv.y);
        hv[2] = (__bf16)tanhf(ev.z + pv.z);
        hv[3] = (__bf16)tanhf(ev.w + pv.w);
        *(v4bf*)(hA + (size_t)u * kHPad + k4) = hv;
    }

    v8f acc[8];
#pragma unroll
    for (int i = 0; i < 8; ++i) acc[i] = (v8f){0.f, 0.f, 0.f, 0.f, 0.f, 0.f, 0.f, 0.f};

    // CDNA5 16-bit fragment K selection per the ISA VGPR layout tables:
    //   A (16x32): lanes 0-15 hold K {0..7, 16..23}; lanes 16-31 hold K {8..15, 24..31}
    //   B (32x16): lanes 0-15 hold K {0..7, 16..23}; lanes 16-31 hold K {8..15, 24..31}
    const int ksel = lo ? 0 : 8;    // A: first 8-elem segment offset
    const int koff = lo ? 0 : 8;    // B: first 8-elem segment offset

    // W2 staging: each thread owns row (n0+tid), kKC=64 bf16 per step = 8 x b128
    const __bf16* w2row = w2bf + (size_t)(n0 + tid) * kH;
    __bf16* dstRow = bL + (size_t)tid * kBPad;

    v8bf stage[8];
#pragma unroll
    for (int i = 0; i < 8; ++i) stage[i] = *(const v8bf*)(w2row + 8 * i);   // prefetch kk=0

    for (int kk = 0; kk < kH; kk += kKC) {
        // commit staged chunk to LDS (prev iter's readers passed the tail barrier)
#pragma unroll
        for (int i = 0; i < 8; ++i) *(v8bf*)(dstRow + 8 * i) = stage[i];
        __syncthreads();

        // kick off next chunk's global loads; they overlap the WMMA block below
        if (kk + kKC < kH) {
            const __bf16* src = w2row + kk + kKC;
#pragma unroll
            for (int i = 0; i < 8; ++i) stage[i] = *(const v8bf*)(src + 8 * i);
        }

        // two K=32 sub-steps per staged chunk
#pragma unroll
        for (int s = 0; s < 2; ++s) {
            const int ko = s * 32;

            // A fragment (16 u-rows x 32 K) from LDS
            union { v16bf v; v8bf h2[2]; } af;
            const __bf16* abase = hA + (size_t)(mi * 16 + lrow) * kHPad + kk + ko + ksel;
            af.h2[0] = *(const v8bf*)(abase);
            af.h2[1] = *(const v8bf*)(abase + 16);

            // batch all 8 B fragments (one ds_load clause, one wait) ...
            union { v16bf v; v8bf h2[2]; } bfrag[8];
#pragma unroll
            for (int nt = 0; nt < 8; ++nt) {
                const __bf16* bbase =
                    bL + (size_t)(nj * 128 + nt * 16 + lrow) * kBPad + ko + koff;
                bfrag[nt].h2[0] = *(const v8bf*)(bbase);
                bfrag[nt].h2[1] = *(const v8bf*)(bbase + 16);
            }
            // ... then 8 back-to-back WMMAs (distinct accumulators, no RAW hazard)
#pragma unroll
            for (int nt = 0; nt < 8; ++nt) {
                acc[nt] = __builtin_amdgcn_wmma_f32_16x16x32_bf16(
                    false, af.v, false, bfrag[nt].v, (short)0, acc[nt], false, false);
            }
        }
        __syncthreads();
    }

    // ---- Epilogue: +b2 and store fp32 (coalesced 16-lane rows) ----
    float* obase = out + (size_t)bt * kU * kV;
#pragma unroll
    for (int nt = 0; nt < 8; ++nt) {
        int vcol = n0 + nj * 128 + nt * 16 + lrow;
        float bias = b2[vcol];
#pragma unroll
        for (int r = 0; r < 8; ++r) {
            int urow = mi * 16 + r + (lo ? 0 : 8);   // C layout: VGPR r -> M=r (+8 hi lanes)
            obase[(size_t)urow * kV + vcol] = acc[nt][r] + bias;
        }
    }
}

// ---------------------------------------------------------------------------
extern "C" void kernel_launch(void* const* d_in, const int* in_sizes, int n_in,
                              void* d_out, int out_size, void* d_ws, size_t ws_size,
                              hipStream_t stream)
{
    const float* enc  = (const float*)d_in[0];   // (B,T,D)
    const float* pred = (const float*)d_in[1];   // (B,U,D)
    const float* w1   = (const float*)d_in[2];   // (H,D)
    const float* b1   = (const float*)d_in[3];   // (H)
    const float* w2   = (const float*)d_in[4];   // (V,H)
    const float* b2   = (const float*)d_in[5];   // (V)
    float* out = (float*)d_out;                  // (B,T,U,V)

    // Workspace layout: E1 (B*T x H f32) | P1 (B*U x H f32) | w2bf (V*H bf16)
    float*  E1   = (float*)d_ws;
    float*  P1   = E1 + (size_t)kB * kT * kH;
    __bf16* w2bf = (__bf16*)(P1 + (size_t)kB * kU * kH);

    // Prep: tiny projections (1.3 GFLOP) + weight conversion
    int projElems = (kB * kT + kB * kU) * kH;    // 2560 * 512
    jn_prep_proj<<<dim3(projElems / 256), dim3(256), 0, stream>>>(enc, pred, w1, b1, E1, P1);
    jn_cvt_w2<<<dim3((kV * kH) / 256), dim3(256), 0, stream>>>(w2, w2bf);

    // Main fused tanh + GEMM2 (68.7 GFLOP bf16 WMMA)
    size_t shmem = ((size_t)kU * kHPad + (size_t)kNT * kBPad) * sizeof(__bf16); // 103,424 B
    jn_main<<<dim3(kB * kT, kV / kNT), dim3(256), shmem, stream>>>(E1, P1, w2bf, b2, out);
}
